// Attention_26113401160232
// MI455X (gfx1250) — matrix operand order repaired and checked by reading, not compiled, not run on hardware
//
#include <hip/hip_runtime.h>
#include <hip/hip_bf16.h>

// ---------------------------------------------------------------------------
// Types for gfx1250 WMMA (wave32): v_wmma_f32_16x16x32_bf16
// ---------------------------------------------------------------------------
typedef __attribute__((ext_vector_type(16))) __bf16 v16bf;
typedef __attribute__((ext_vector_type(8)))  __bf16 v8bf;
typedef __attribute__((ext_vector_type(8)))  float  v8f;
typedef __attribute__((ext_vector_type(4)))  float  v4f;

union V16U { v16bf v; v8bf h[2]; };

#define BM 128
#define BN 128
#define BK 32

// ---------------------------------------------------------------------------
// Tiled bf16 WMMA GEMM:  C[M,N] = scale * (A[M,K] @ B + bias)
//  - A_F32 / B_F32 : operand is fp32 in global (VALU-converted on LDS staging)
//  - B_NT          : B supplied row-major as [N,K] (QK^T); otherwise [K,N]
//  - OUT_F32       : write fp32, else bf16
//
// 128 threads = 4 wave32 (2x2), each wave owns a 64x64 patch = 4x4 WMMA tiles
// (16 v_wmma per k-step per wave vs 16 ds_load_b128 fragment reads).
// LDS double buffer; bf16 straight-copy tiles are staged with
// GLOBAL_LOAD_ASYNC_TO_LDS_B128 (ASYNCcnt) so the DMA overlaps the WMMAs.
// ---------------------------------------------------------------------------
template <bool A_F32, bool B_F32, bool B_NT, bool HAS_BIAS, bool OUT_F32>
__global__ __launch_bounds__(128) void gemm_bf16_wmma(
    const void* __restrict__ Ap, const void* __restrict__ Bp,
    const float* __restrict__ bias, void* __restrict__ Cp,
    int K, int lda, int ldb, int ldc, float scale)
{
    __shared__ __align__(16) __bf16 lsA[2][BM * BK];
    __shared__ __align__(16) __bf16 lsB[2][BN * BK];

    const int tid  = threadIdx.x;
    const int lane = tid & 31;
    const int wid  = tid >> 5;          // 0..3
    const int wm   = wid >> 1;          // 0..1 : 64-row slab
    const int wn   = wid & 1;           // 0..1 : 64-col slab
    const long blockM = (long)blockIdx.y * BM;
    const long blockN = (long)blockIdx.x * BN;

    const int l15  = lane & 15;
    const int half = lane >> 4;         // 0 or 1

    v8f acc[4][4];
#pragma unroll
    for (int mi = 0; mi < 4; ++mi)
#pragma unroll
        for (int ni = 0; ni < 4; ++ni)
            acc[mi][ni] = (v8f){0.f, 0.f, 0.f, 0.f, 0.f, 0.f, 0.f, 0.f};

    // ---- staging: each thread owns one full 32-element k-row of its tile ----
    auto stageA = [&](int kt, __bf16* __restrict__ dst) {
        if constexpr (A_F32) {
            const float* Ag = (const float*)Ap + (blockM + tid) * (long)lda + kt;
#pragma unroll
            for (int i = 0; i < 32; i += 4) {
                v4f t = *(const v4f*)(Ag + i);
                dst[tid * BK + i + 0] = (__bf16)t.x;
                dst[tid * BK + i + 1] = (__bf16)t.y;
                dst[tid * BK + i + 2] = (__bf16)t.z;
                dst[tid * BK + i + 3] = (__bf16)t.w;
            }
        } else {
            const __bf16* Ag = (const __bf16*)Ap + (blockM + tid) * (long)lda + kt;
#pragma unroll
            for (int i = 0; i < 4; ++i) {
                unsigned ldsa = (unsigned)(unsigned long long)(uintptr_t)&dst[tid * BK + i * 8];
                unsigned long long ga = (unsigned long long)(uintptr_t)(Ag + i * 8);
                asm volatile("global_load_async_to_lds_b128 %0, %1, off"
                             :: "v"(ldsa), "v"(ga) : "memory");
            }
        }
    };

    auto stageB = [&](int kt, __bf16* __restrict__ dst) {
        if constexpr (B_NT) {
            // B is [N,K] row-major: rows are output columns -> straight copy
            if constexpr (B_F32) {
                const float* Bg = (const float*)Bp + (blockN + tid) * (long)ldb + kt;
#pragma unroll
                for (int i = 0; i < 32; i += 4) {
                    v4f t = *(const v4f*)(Bg + i);
                    dst[tid * BK + i + 0] = (__bf16)t.x;
                    dst[tid * BK + i + 1] = (__bf16)t.y;
                    dst[tid * BK + i + 2] = (__bf16)t.z;
                    dst[tid * BK + i + 3] = (__bf16)t.w;
                }
            } else {
                const __bf16* Bg = (const __bf16*)Bp + (blockN + tid) * (long)ldb + kt;
#pragma unroll
                for (int i = 0; i < 4; ++i) {
                    unsigned ldsa = (unsigned)(unsigned long long)(uintptr_t)&dst[tid * BK + i * 8];
                    unsigned long long ga = (unsigned long long)(uintptr_t)(Bg + i * 8);
                    asm volatile("global_load_async_to_lds_b128 %0, %1, off"
                                 :: "v"(ldsa), "v"(ga) : "memory");
                }
            }
        } else {
            // B is [K,N] row-major: transpose into [n][k] on the store side
            const int kk = tid & 31;           // k row
            const int ns = (tid >> 5) * 32;    // n segment
            if constexpr (B_F32) {
                const float* Bg = (const float*)Bp + (long)(kt + kk) * ldb + blockN + ns;
#pragma unroll
                for (int i = 0; i < 32; i += 4) {
                    v4f t = *(const v4f*)(Bg + i);
                    dst[(ns + i + 0) * BK + kk] = (__bf16)t.x;
                    dst[(ns + i + 1) * BK + kk] = (__bf16)t.y;
                    dst[(ns + i + 2) * BK + kk] = (__bf16)t.z;
                    dst[(ns + i + 3) * BK + kk] = (__bf16)t.w;
                }
            } else {
                const __bf16* Bg = (const __bf16*)Bp + (long)(kt + kk) * ldb + blockN + ns;
#pragma unroll
                for (int i = 0; i < 32; i += 8) {
                    v8bf t = *(const v8bf*)(Bg + i);
#pragma unroll
                    for (int j = 0; j < 8; ++j)
                        dst[(ns + i + j) * BK + kk] = t[j];
                }
            }
        }
    };

    auto compute = [&](const __bf16* __restrict__ bufA, const __bf16* __restrict__ bufB) {
        V16U afr[4], bfr[4];
#pragma unroll
        for (int mi = 0; mi < 4; ++mi) {
            // A 16x32 bf16: lanes 0-15 -> K {0..7,16..23}; lanes 16-31 -> K {8..15,24..31}
            const __bf16* base = bufA + (wm * 64 + mi * 16 + l15) * BK;
            afr[mi].h[0] = *(const v8bf*)(base + half * 8);
            afr[mi].h[1] = *(const v8bf*)(base + half * 8 + 16);
        }
#pragma unroll
        for (int ni = 0; ni < 4; ++ni) {
            // B 32x16 bf16: lane = N; lanes 0-15 -> K {0..7,16..23}; lanes 16-31 -> K {8..15,24..31}
            const __bf16* base = bufB + (wn * 64 + ni * 16 + l15) * BK;
            bfr[ni].h[0] = *(const v8bf*)(base + half * 8);
            bfr[ni].h[1] = *(const v8bf*)(base + half * 8 + 16);
        }
#pragma unroll
        for (int mi = 0; mi < 4; ++mi)
#pragma unroll
            for (int ni = 0; ni < 4; ++ni)
                acc[mi][ni] = __builtin_amdgcn_wmma_f32_16x16x32_bf16(
                    false, afr[mi].v, false, bfr[ni].v,
                    (short)0, acc[mi][ni], false, false);
    };

    constexpr bool ASYNC = (!A_F32) || (!B_F32 && B_NT);

    // ---- ping-pong main loop: DMA of tile t+1 overlaps WMMAs of tile t ----
    int p = 0;
    stageA(0, lsA[0]);
    stageB(0, lsB[0]);
    if constexpr (ASYNC) asm volatile("s_wait_asynccnt 0x0" ::: "memory");
    __syncthreads();

    for (int kt = 0; kt < K; kt += BK) {
        const int nxt = kt + BK;
        if (nxt < K) {
            stageA(nxt, lsA[p ^ 1]);
            stageB(nxt, lsB[p ^ 1]);
        }
        compute(lsA[p], lsB[p]);
        if constexpr (ASYNC) asm volatile("s_wait_asynccnt 0x0" ::: "memory");
        __syncthreads();
        p ^= 1;
    }

    // ---- epilogue: C layout -> (M = r + 8*(lane>=16), N = lane&15) ----
    const int mhi = half * 8;
#pragma unroll
    for (int mi = 0; mi < 4; ++mi)
#pragma unroll
        for (int ni = 0; ni < 4; ++ni)
#pragma unroll
            for (int r = 0; r < 8; ++r) {
                long row = blockM + wm * 64 + mi * 16 + mhi + r;
                long col = blockN + wn * 64 + ni * 16 + l15;
                float v = acc[mi][ni][r];
                if constexpr (HAS_BIAS) v += bias[col];
                v *= scale;
                if constexpr (OUT_F32)
                    ((float*)Cp)[row * (long)ldc + col] = v;
                else
                    ((__bf16*)Cp)[row * (long)ldc + col] = (__bf16)v;
            }
}

// ---------------------------------------------------------------------------
// Row softmax over 2048 fp32 values; writes probabilities in place as bf16
// packed at the start of each row's own fp32 storage (bf16 row stride = 2*ncols).
// One 256-thread workgroup per row; every input element is register-resident
// before any aliasing store happens -> race-free in-place conversion.
// ---------------------------------------------------------------------------
__global__ __launch_bounds__(256) void softmax_rows(float* __restrict__ scores, int ncols)
{
    __shared__ float red[256];
    const int tid = threadIdx.x;
    float* row = scores + (size_t)blockIdx.x * ncols;
    __bf16* outp = (__bf16*)scores + (size_t)blockIdx.x * (2 * ncols);

    float vals[8];
    float m = -3.0e38f;
#pragma unroll
    for (int i = 0; i < 8; ++i) {
        vals[i] = row[tid + i * 256];
        m = fmaxf(m, vals[i]);
    }
    red[tid] = m;
    __syncthreads();
    for (int s = 128; s > 0; s >>= 1) {
        if (tid < s) red[tid] = fmaxf(red[tid], red[tid + s]);
        __syncthreads();
    }
    m = red[0];
    __syncthreads();

    float sum = 0.f;
#pragma unroll
    for (int i = 0; i < 8; ++i) {
        vals[i] = __expf(vals[i] - m);
        sum += vals[i];
    }
    red[tid] = sum;
    __syncthreads();
    for (int s = 128; s > 0; s >>= 1) {
        if (tid < s) red[tid] += red[tid + s];
        __syncthreads();
    }
    const float inv = 1.0f / red[0];
#pragma unroll
    for (int i = 0; i < 8; ++i)
        outp[tid + i * 256] = (__bf16)(vals[i] * inv);
}

// ---------------------------------------------------------------------------
// Host launcher
//   inputs: query, key, value, Wq, bq, Wk, bk, Wv, bv (all fp32)
//   out: [4, 2048, 1024] fp32
//   ws: 48 MB bf16 q/k/v + 16 MB reusable per-batch score buffer = 64 MB
// ---------------------------------------------------------------------------
extern "C" void kernel_launch(void* const* d_in, const int* in_sizes, int n_in,
                              void* d_out, int out_size, void* d_ws, size_t ws_size,
                              hipStream_t stream)
{
    (void)in_sizes; (void)n_in; (void)out_size; (void)ws_size;

    const float* query = (const float*)d_in[0];
    const float* key   = (const float*)d_in[1];
    const float* value = (const float*)d_in[2];
    const float* Wq    = (const float*)d_in[3];
    const float* bq    = (const float*)d_in[4];
    const float* Wk    = (const float*)d_in[5];
    const float* bk    = (const float*)d_in[6];
    const float* Wv    = (const float*)d_in[7];
    const float* bv    = (const float*)d_in[8];
    float* out = (float*)d_out;

    const int B = 4, S = 2048, D = 1024;
    const float BETA = 0.03125f;   // 1/sqrt(1024), folded into q projection

    char* ws = (char*)d_ws;
    const size_t SLAB = (size_t)B * S * D * 2;   // 16 MB bf16
    char* qb = ws;
    char* kb = ws + SLAB;
    char* vb = ws + 2 * SLAB;
    float* scores = (float*)(ws + 3 * SLAB);     // 2048*2048 fp32, reused per batch

    dim3 blk(128);

    // Projections (all batches fused as M = B*S = 8192), epilogue writes bf16.
    gemm_bf16_wmma<true, true, false, true, false>
        <<<dim3(D / BN, (B * S) / BM), blk, 0, stream>>>(query, Wq, bq, qb, D, D, D, D, BETA);
    gemm_bf16_wmma<true, true, false, true, false>
        <<<dim3(D / BN, (B * S) / BM), blk, 0, stream>>>(key, Wk, bk, kb, D, D, D, D, 1.0f);
    gemm_bf16_wmma<true, true, false, true, false>
        <<<dim3(D / BN, (B * S) / BM), blk, 0, stream>>>(value, Wv, bv, vb, D, D, D, D, 1.0f);

    for (int b = 0; b < B; ++b) {
        const void* qbb = qb + (size_t)b * S * D * 2;
        const void* kbb = kb + (size_t)b * S * D * 2;
        const void* vbb = vb + (size_t)b * S * D * 2;
        float* ob = out + (size_t)b * S * D;

        // scores = (beta*q) @ k^T   [S,S] fp32   (both operands async-staged)
        gemm_bf16_wmma<false, false, true, false, true>
            <<<dim3(S / BN, S / BM), blk, 0, stream>>>(qbb, kbb, nullptr, scores,
                                                       D, D, D, S, 1.0f);
        // softmax rows, in-place fp32 -> bf16 (bf16 row stride 2*S)
        softmax_rows<<<S, dim3(256), 0, stream>>>(scores, S);

        // out = P @ v   [S,D] fp32 ; P is bf16 with lda = 2*S (A async-staged)
        gemm_bf16_wmma<false, false, false, false, true>
            <<<dim3(D / BN, S / BM), blk, 0, stream>>>((const void*)scores, vbb, nullptr, ob,
                                                       S, 2 * S, D, D, 1.0f);
    }
}
